// MultipleMappings_5952824672291
// MI455X (gfx1250) — compile-verified, matrix operand order checked by reading, not run on hardware
//
#include <hip/hip_runtime.h>

// Problem constants (from reference): B=64, S=4096, D=E=256, L=16
#define B_ 64
#define S_ 4096
#define D_ 256
#define L_ 16

#define SM_TILE 256      // S rows per workgroup
#define THREADS 512      // 16 waves (wave32)

typedef __attribute__((ext_vector_type(16))) __bf16 v16bf;
typedef __attribute__((ext_vector_type(8)))  float  v8f;
typedef __attribute__((ext_vector_type(4)))  float  v4f;   // NT-builtin-compatible float4

// LDS layout (256KB): kt-major, hi/lo interleaved per fragment so the whole inner
// loop addresses from ONE per-kt lane base with 16-bit DS immediate offsets:
//   fragment(kt, nt) at byte (kt*16 + nt)*2048 ; hi half at +0, lo half at +1024
//   within a half: lane*32 bytes, 16 bf16 per lane.
// Per-lane element mapping (matches WMMA 16-bit A/B VGPR layout): lane L holds
// column n = nt*16 + (L&15); value j maps to k = kt*32 + ((L&16)?8:0) + (j<8 ? j : j+8).

__global__ __launch_bounds__(THREADS, 1)
void mm_bf16split_kernel(const float* __restrict__ right_emb,
                         const float* __restrict__ mapping,
                         const int*   __restrict__ pair_id,
                         float*       __restrict__ out)
{
    extern __shared__ char smem_raw[];
    __bf16* lds = reinterpret_cast<__bf16*>(smem_raw);

    const int b     = blockIdx.y;
    const int stile = blockIdx.x;
    const int tid   = threadIdx.x;
    const int lane  = tid & 31;
    const int wave  = tid >> 5;

    const float* Asrc = right_emb + ((size_t)b * S_ + (size_t)stile * SM_TILE) * D_;
    float*       Cdst = out       + ((size_t)b * S_ + (size_t)stile * SM_TILE) * D_;

    const int lang = pair_id[b];

    if (lang < 0) {
        // Passthrough: out = right_emb for this slab (256x256 f32). Streaming -> NT both ways.
        const v4f* src4 = reinterpret_cast<const v4f*>(Asrc);
        v4f*       dst4 = reinterpret_cast<v4f*>(Cdst);
        #pragma unroll 4
        for (int i = tid; i < (SM_TILE * D_) / 4; i += THREADS) {
            v4f v = __builtin_nontemporal_load(&src4[i]);
            __builtin_nontemporal_store(v, &dst4[i]);
        }
        return;
    }

    // ---- Stage W = mapping[lang] (256x256 f32) into LDS bf16 hi/lo fragments ----
    // W is reused by many blocks: default (RT) hints so it stays L2-resident.
    const float* W = mapping + (size_t)lang * D_ * D_;

    for (int slot = tid; slot < 128 * 32; slot += THREADS) {   // 128 fragments x 32 lanes
        const int f  = slot >> 5;      // fragment id = kt*16 + nt
        const int ln = slot & 31;
        const int kt = f >> 4;
        const int nt = f & 15;
        const int n     = nt * 16 + (ln & 15);
        const int kbase = kt * 32 + ((ln & 16) ? 8 : 0);

        const float* wrow = W + (unsigned)n * D_;
        const v4f* p0 = reinterpret_cast<const v4f*>(wrow + kbase);
        const v4f* p1 = reinterpret_cast<const v4f*>(wrow + kbase + 16);
        v4f q0 = p0[0], q1 = p0[1];   // k = kbase .. kbase+7
        v4f q2 = p1[0], q3 = p1[1];   // k = kbase+16 .. kbase+23
        float wv[16] = { q0.x, q0.y, q0.z, q0.w, q1.x, q1.y, q1.z, q1.w,
                         q2.x, q2.y, q2.z, q2.w, q3.x, q3.y, q3.z, q3.w };

        __bf16* hi = lds + (unsigned)f * 1024 + ln * 16;   // fragment pair = 1024 elems (2048B)
        __bf16* lo = hi + 512;                             // lo half at +1024B
        #pragma unroll
        for (int j = 0; j < 16; ++j) {
            float x = wv[j];
            __bf16 h = (__bf16)x;
            hi[j] = h;
            lo[j] = (__bf16)(x - (float)h);
        }
    }
    __syncthreads();

    // ---- Compute: each wave -> 16 rows x full N=256 (16 accumulator tiles) ----
    const int m0  = wave * 16;
    const int row = m0 + (lane & 15);
    const float* arow = Asrc + (unsigned)row * D_;
    const int klane = (lane & 16) ? 8 : 0;

    v8f acc[16];
    #pragma unroll
    for (int i = 0; i < 16; ++i)
        acc[i] = (v8f){0.f, 0.f, 0.f, 0.f, 0.f, 0.f, 0.f, 0.f};

    // Prefetch raw A for kt=0 (non-temporal: A is streamed exactly once).
    v4f q0, q1, q2, q3;
    {
        const v4f* p0 = reinterpret_cast<const v4f*>(arow + klane);
        const v4f* p1 = reinterpret_cast<const v4f*>(arow + klane + 16);
        q0 = __builtin_nontemporal_load(&p0[0]);
        q1 = __builtin_nontemporal_load(&p0[1]);
        q2 = __builtin_nontemporal_load(&p1[0]);
        q3 = __builtin_nontemporal_load(&p1[1]);
    }

    #pragma unroll 1
    for (int kt = 0; kt < 8; ++kt) {
        // Issue next-kt A prefetch early so global latency hides under this kt's WMMAs.
        v4f n0, n1, n2, n3;
        if (kt < 7) {
            const int kb2 = (kt + 1) * 32 + klane;
            const v4f* p0 = reinterpret_cast<const v4f*>(arow + kb2);
            const v4f* p1 = reinterpret_cast<const v4f*>(arow + kb2 + 16);
            n0 = __builtin_nontemporal_load(&p0[0]);
            n1 = __builtin_nontemporal_load(&p0[1]);
            n2 = __builtin_nontemporal_load(&p1[0]);
            n3 = __builtin_nontemporal_load(&p1[1]);
        }

        // Convert current raw A to bf16 hi/lo fragments.
        float av[16] = { q0.x, q0.y, q0.z, q0.w, q1.x, q1.y, q1.z, q1.w,
                         q2.x, q2.y, q2.z, q2.w, q3.x, q3.y, q3.z, q3.w };
        v16bf a_hi, a_lo;
        #pragma unroll
        for (int j = 0; j < 16; ++j) {
            float x = av[j];
            __bf16 h = (__bf16)x;
            a_hi[j] = h;
            a_lo[j] = (__bf16)(x - (float)h);
        }

        // One lane base per kt: all 32 fragment halves reachable via 16-bit DS offsets.
        const __bf16* ktbase = lds + (unsigned)kt * 16384 + (unsigned)lane * 16;

        // Branch-free B double buffer (next = (nt+1)&15) so the rotation stays in SSA
        // and the scheduler can overlap ds_loads with WMMAs using partial dscnt waits.
        v16bf b_hi = *reinterpret_cast<const v16bf*>(ktbase);
        v16bf b_lo = *reinterpret_cast<const v16bf*>(ktbase + 512);

        #pragma unroll
        for (int nt = 0; nt < 16; ++nt) {
            const unsigned nxt = ((unsigned)(nt + 1) & 15u) * 1024u;
            v16bf bh_n = *reinterpret_cast<const v16bf*>(ktbase + nxt);
            v16bf bl_n = *reinterpret_cast<const v16bf*>(ktbase + nxt + 512);
            // hi*hi + lo*hi + hi*lo : fp32-class precision at 3x bf16 WMMA cost
            acc[nt] = __builtin_amdgcn_wmma_f32_16x16x32_bf16(
                false, a_hi, false, b_hi, (short)0, acc[nt], false, false);
            acc[nt] = __builtin_amdgcn_wmma_f32_16x16x32_bf16(
                false, a_lo, false, b_hi, (short)0, acc[nt], false, false);
            acc[nt] = __builtin_amdgcn_wmma_f32_16x16x32_bf16(
                false, a_hi, false, b_lo, (short)0, acc[nt], false, false);
            b_hi = bh_n;
            b_lo = bl_n;
        }

        q0 = n0; q1 = n1; q2 = n2; q3 = n3;
    }

    // ---- Store C (non-temporal: output is written exactly once, keep L2 for W).
    //      32-bit C/D layout: lanes 0-15 -> N=lane, VGPR r -> M=r;
    //      lanes 16-31 -> N=lane-16, VGPR r -> M=8+r.
    float* cbase = Cdst + (unsigned)(m0 + ((lane & 16) ? 8 : 0)) * D_ + (lane & 15);
    #pragma unroll
    for (int nt = 0; nt < 16; ++nt) {
        #pragma unroll
        for (int r = 0; r < 8; ++r) {
            __builtin_nontemporal_store(acc[nt][r], &cbase[(unsigned)r * D_ + nt * 16]);
        }
    }
}

extern "C" void kernel_launch(void* const* d_in, const int* in_sizes, int n_in,
                              void* d_out, int out_size, void* d_ws, size_t ws_size,
                              hipStream_t stream) {
    const float* right_emb = (const float*)d_in[0];   // (B, S, D) f32
    const float* mapping   = (const float*)d_in[1];   // (L, D, D) f32
    const int*   pair_id   = (const int*)d_in[2];     // (B, 1) i32
    float*       out       = (float*)d_out;           // (B, S, D) f32

    dim3 grid(S_ / SM_TILE, B_);   // (16, 64)
    dim3 block(THREADS);
    size_t lds_bytes = 2u * 128u * 32u * 16u * sizeof(__bf16);  // 262144 B = 256 KB
    mm_bf16split_kernel<<<grid, block, lds_bytes, stream>>>(right_emb, mapping, pair_id, out);
}
